// alpha_net_spatial2_14843406975142
// MI455X (gfx1250) — matrix operand order repaired and checked by reading, not compiled, not run on hardware
//
#include <hip/hip_runtime.h>
#include <hip/hip_fp16.h>

typedef __attribute__((ext_vector_type(16))) _Float16 v16h;
typedef __attribute__((ext_vector_type(8)))  _Float16 v8h;
typedef __attribute__((ext_vector_type(8)))  float    v8f;

#define Bb 16
#define Cc 256
#define Nn 1024
#define CHUNKS 32   // Nn / 32  (K chunks of 32 for 16x16x32 WMMA)
#define CTILES 16   // Cc / 16
#define ASTRIDE 24  // padded A-fragment LDS row stride in halves (48B: 16B aligned, conflict-reduced)
#define ABUF (32 * ASTRIDE)

// exp(tanh(x)) with fast hardware ops only:
// tanh(x) = 1 - 2/(e^{2x}+1) using v_rcp_f32 (not IEEE divide); saturates correctly at +-inf.
__device__ __forceinline__ float exp_tanh(float x) {
    float e  = __expf(2.0f * x);
    float th = 1.0f - 2.0f * __builtin_amdgcn_rcpf(e + 1.0f);
    return __expf(th);
}

// ---------------- Pass 1: per-position logits a,b (dot over C) ----------------
__global__ __launch_bounds__(256)
void k_ab(const float* __restrict__ h, const float* __restrict__ w1w,
          const float* __restrict__ w1b, const float* __restrict__ w2w,
          const float* __restrict__ w2b, float* __restrict__ aq, float* __restrict__ bq) {
    int idx = blockIdx.x * 256 + threadIdx.x;          // b*Nn + n
    int b = idx >> 10, n = idx & (Nn - 1);
    const float* hp = h + (size_t)b * Cc * Nn + n;
    float sa = 0.f, sb = 0.f;
#pragma unroll 4
    for (int c = 0; c < Cc; ++c) {
        float v = hp[(size_t)c * Nn];                  // consecutive threads -> consecutive n: coalesced
        sa = fmaf(v, w1w[c], sa);
        sb = fmaf(v, w2w[c], sb);
    }
    aq[idx] = sa + w1b[0];
    bq[idx] = sb + w2b[0];
}

// ---------------- Pass 2: pack hf into f16 B-fragment-ready layout ----------------
// hfB flat index = ((((b*32 + chunk)*16 + ctile)*32 + lane)*16 + e)
//   value = hf16[j][c], j = chunk*32 + ((e&8)?16:0) + ((lane>=16)?8:0) + (e&7), c = ctile*16 + (lane&15)
// matches ISA wave32 B-matrix (32x16, 16-bit): lanes 0-15 K=0..7,16..23, lanes 16-31 K=8..15,24..31, N=lane&15.
__global__ __launch_bounds__(256)
void k_pack(const float* __restrict__ h, _Float16* __restrict__ hfB) {
    unsigned idx  = blockIdx.x * 256u + threadIdx.x;
    unsigned e    = idx & 15u;
    unsigned lane = (idx >> 4) & 31u;
    unsigned ct   = (idx >> 9) & 15u;
    unsigned ch   = (idx >> 13) & 31u;
    unsigned b    = idx >> 18;
    unsigned j    = ch * 32u + ((e & 8u) << 1) + ((lane >> 4) << 3) + (e & 7u);
    unsigned c    = ct * 16u + (lane & 15u);
    hfB[idx] = (_Float16)h[((size_t)b * Cc + c) * Nn + j];   // writes coalesced; reads L2-resident
}

// ---------------- Main: fused score-gen + softmax-normalized WMMA GEMM ----------------
__global__ __launch_bounds__(256)
void k_attn(const _Float16* __restrict__ hfB, const float* __restrict__ aq,
            const float* __restrict__ bq, float* __restrict__ out) {
    __shared__ _Float16 Afrag[2 * ABUF];       // double-buffered 16x32 f16 score tile, A-fragment layout
    __shared__ float    bq_s[Nn];              // all key logits for this batch
    __shared__ float    sums[16];              // softmax denominators per row
    __shared__ float    obuf[Cc * 17];         // output staging (stride 17 kills bank conflicts)

    const int t     = threadIdx.x;
    const int b     = blockIdx.x >> 6;
    const int ibase = (blockIdx.x & 63) << 4;

    for (int k = t; k < Nn; k += 256) bq_s[k] = bq[b * Nn + k];
    if (t < 16) sums[t] = 0.f;

    const int i_loc = t & 15;
    const int jrow  = t >> 4;                              // 0..15
    const float av  = aq[b * Nn + ibase + i_loc];
    // A-fragment scatter target for (i_loc, jrow) and (i_loc, jrow+16):
    // lane = i + 16*((j>>3)&1); elem e = (j&7) + 8*(j>>4)  ->  e and e+8, same lane
    const int aw_off = (i_loc + (((jrow >> 3) & 1) << 4)) * ASTRIDE + (jrow & 7);

    const int wave = t >> 5, lane = t & 31;
    const _Float16* bbase = hfB + (size_t)b * (CHUNKS * CTILES * 32 * 16) + lane * 16;

    v8f acc0 = {}; v8f acc1 = {};
    float rsum = 0.f;
    __syncthreads();

    for (int ch = 0; ch < CHUNKS; ++ch) {
        const int buf = (ch & 1) * ABUF;
        int j0 = ch * 32 + jrow;
        float s0 = exp_tanh(av + bq_s[j0]);
        float s1 = exp_tanh(av + bq_s[j0 + 16]);
        rsum += s0 + s1;
        Afrag[buf + aw_off]     = (_Float16)s0;
        Afrag[buf + aw_off + 8] = (_Float16)s1;
        __syncthreads();   // single barrier: buffer (ch&1) complete; next iter writes the other buffer

        const v8h* ap = (const v8h*)&Afrag[buf + lane * ASTRIDE];
        v8h a_lo = ap[0], a_hi = ap[1];
        v16h afr = __builtin_shufflevector(a_lo, a_hi,
                    0,1,2,3,4,5,6,7,8,9,10,11,12,13,14,15);
        const _Float16* bp0 = bbase + (size_t)(ch * CTILES + wave) * 512;
        const _Float16* bp1 = bp0 + 8 * 512;
        __builtin_prefetch(bp0 + CTILES * 512, 0, 3);      // unconditional: no branch in hot loop
        v16h bf0 = *(const v16h*)bp0;                      // 32B coalesced, fragment-ready
        v16h bf1 = *(const v16h*)bp1;
        acc0 = __builtin_amdgcn_wmma_f32_16x16x32_f16(false, afr, false, bf0,
                                                      (short)0, acc0, false, false);
        acc1 = __builtin_amdgcn_wmma_f32_16x16x32_f16(false, afr, false, bf1,
                                                      (short)0, acc1, false, false);
    }

    atomicAdd(&sums[i_loc], rsum);                         // ds_add_f32
    __syncthreads();

    // C/D layout: VGPR r <-> M = r + 8*(lane>=16); N = lane&15. Normalize + leaky ReLU, stage to LDS.
    const int moff = (lane >> 4) << 3;
    const int ccol = lane & 15;
#pragma unroll
    for (int r = 0; r < 8; ++r) {
        float inv = __builtin_amdgcn_rcpf(sums[moff + r]);
        float v0 = acc0[r] * inv;  v0 = v0 > 0.f ? v0 : 0.2f * v0;
        float v1 = acc1[r] * inv;  v1 = v1 > 0.f ? v1 : 0.2f * v1;
        obuf[(wave * 16 + ccol) * 17 + moff + r]         = v0;
        obuf[((wave + 8) * 16 + ccol) * 17 + moff + r]   = v1;
    }
    __syncthreads();

    // Store: groups of 4 lanes emit one 64B-contiguous c-row (4 x float4) -> full-width transactions.
#pragma unroll
    for (int rr = 0; rr < 4; ++rr) {
        int p  = t + 256 * rr;        // 0..1023 float4 pieces
        int c  = p >> 2;
        int io = (p & 3) << 2;
        float4 v = { obuf[c * 17 + io],     obuf[c * 17 + io + 1],
                     obuf[c * 17 + io + 2], obuf[c * 17 + io + 3] };
        *(float4*)&out[((size_t)b * Cc + c) * Nn + ibase + io] = v;
    }
}

extern "C" void kernel_launch(void* const* d_in, const int* in_sizes, int n_in,
                              void* d_out, int out_size, void* d_ws, size_t ws_size,
                              hipStream_t stream) {
    const float* h   = (const float*)d_in[0];
    const float* w1w = (const float*)d_in[1];
    const float* w1b = (const float*)d_in[2];
    const float* w2w = (const float*)d_in[3];
    const float* w2b = (const float*)d_in[4];
    float* out = (float*)d_out;

    char* ws = (char*)d_ws;
    float*    aq  = (float*)ws;                              // 16*1024 f32 = 64KB
    float*    bq  = (float*)(ws + 64 * 1024);                // 64KB
    _Float16* hfB = (_Float16*)(ws + 128 * 1024);            // 16*1024*256 f16 = 8MB

    k_ab  <<<(Bb * Nn) / 256, 256, 0, stream>>>(h, w1w, w1b, w2w, w2b, aq, bq);
    k_pack<<<(Bb * Nn * Cc) / 256, 256, 0, stream>>>(h, hfB);
    k_attn<<<Bb * (Nn / 16), 256, 0, stream>>>(hfB, aq, bq, out);
}
